// Net_274877907022
// MI455X (gfx1250) — compile-verified, matrix operand order checked by reading, not run on hardware
//
#include <hip/hip_runtime.h>
#include <hip/hip_bf16.h>
#include <math.h>

// ---------------------------------------------------------------------------
// Problem constants (match reference)
// ---------------------------------------------------------------------------
#define N_ROWS 8192
#define NODES_ 20000
#define DIN_   768
#define HID_   256
#define M_     4096
#define OUTFE_ 50

typedef __attribute__((ext_vector_type(8)))  __bf16 v8bf;
typedef __attribute__((ext_vector_type(16))) __bf16 v16bf;
typedef __attribute__((ext_vector_type(8)))  float  v8f;

__device__ __forceinline__ unsigned short f32_to_bf16_rne(float f) {
  union { float f; unsigned u; } v; v.f = f;
  unsigned r = v.u + 0x7FFFu + ((v.u >> 16) & 1u);
  return (unsigned short)(r >> 16);
}

// Flat LDS pointer -> 32-bit LDS offset (AS3->flat lowering keeps offset in low dword).
__device__ __forceinline__ unsigned lds_u32(const void* p) {
  return (unsigned)(unsigned long long)(uintptr_t)p;
}

// CDNA5 async global->LDS copy, 16 bytes per active lane. Tracked by ASYNCcnt.
__device__ __forceinline__ void async_copy_b128(unsigned lds_off, const void* gptr) {
  asm volatile("global_load_async_to_lds_b128 %0, %1, off"
               :: "v"(lds_off), "v"((unsigned long long)(uintptr_t)gptr)
               : "memory");
}

__device__ __forceinline__ void wait_async0() {
  asm volatile("s_wait_asynccnt 0x0" ::: "memory");
}

// ---------------------------------------------------------------------------
// Elementwise f32 -> bf16 convert (grid-stride)
// ---------------------------------------------------------------------------
__global__ void conv_f32_to_bf16(const float* __restrict__ src,
                                 unsigned short* __restrict__ dst, long n) {
  long i = (long)blockIdx.x * blockDim.x + threadIdx.x;
  long stride = (long)gridDim.x * blockDim.x;
  for (; i < n; i += stride) dst[i] = f32_to_bf16_rne(src[i]);
}

// Gather rows of AGE_inx by Endx and convert to bf16. One block per output row.
__global__ void gather_conv_bf16(const float* __restrict__ AGE,
                                 const int* __restrict__ idx,
                                 unsigned short* __restrict__ dst) {
  int m = blockIdx.x;
  long srow = (long)idx[m] * DIN_;
  long drow = (long)m * DIN_;
  for (int d = threadIdx.x; d < DIN_; d += blockDim.x)
    dst[drow + d] = f32_to_bf16_rne(AGE[srow + d]);
}

// ---------------------------------------------------------------------------
// Row-wise lintrans: z -> (z-min)/(max-min), then L2 normalize, emit bf16.
// One 256-thread block per row, K = 768.
// ---------------------------------------------------------------------------
__global__ __launch_bounds__(256)
void lintrans_kernel(const float* __restrict__ Z, unsigned short* __restrict__ Eg) {
  const int K = DIN_;
  int row = blockIdx.x;
  int tid = threadIdx.x;
  const float* z = Z + (long)row * K;

  __shared__ float zs[DIN_];
  __shared__ float ra[256];
  __shared__ float rb[256];

  float mn = 3.4e38f, mx = -3.4e38f;
  for (int d = tid; d < K; d += 256) {
    float v = z[d];
    zs[d] = v;
    mn = fminf(mn, v);
    mx = fmaxf(mx, v);
  }
  ra[tid] = mn; rb[tid] = mx;
  __syncthreads();
  for (int s = 128; s > 0; s >>= 1) {
    if (tid < s) {
      ra[tid] = fminf(ra[tid], ra[tid + s]);
      rb[tid] = fmaxf(rb[tid], rb[tid + s]);
    }
    __syncthreads();
  }
  mn = ra[0]; mx = rb[0];
  __syncthreads();

  float inv = 1.0f / (mx - mn);
  float ss = 0.0f;
  for (int d = tid; d < K; d += 256) {
    float v = (zs[d] - mn) * inv;
    zs[d] = v;
    ss += v * v;
  }
  ra[tid] = ss;
  __syncthreads();
  for (int s = 128; s > 0; s >>= 1) {
    if (tid < s) ra[tid] += ra[tid + s];
    __syncthreads();
  }
  float innorm = 1.0f / fmaxf(sqrtf(ra[0]), 1e-12f);
  for (int d = tid; d < K; d += 256)
    Eg[(long)row * K + d] = f32_to_bf16_rne(zs[d] * innorm);
}

// ---------------------------------------------------------------------------
// Tiled bf16 WMMA GEMM:  C[Mrows x Ncols] = epilogue( A[Mrows x K] * B[Ncols x K]^T + bias )
// Block = 256 threads (8 waves). Tile: BM=128, BN=128, BK=32.
// Double-buffered LDS filled by GLOBAL_LOAD_ASYNC_TO_LDS_B128 (ASYNCcnt):
//   preload tile0 -> buf0
//   loop: s_wait_asynccnt 0 ; barrier ; issue tile k+1 -> other buf ; WMMA on cur buf
// Each wave computes a 32x64 patch = 2x4 tiles of v_wmma_f32_16x16x32_bf16.
// FULL=true: tile dims divide Mrows/Ncols exactly -> no bounds guards (straight-line
//            async issue with full EXEC, unguarded epilogue). FULL=false: guarded.
// ---------------------------------------------------------------------------
#define EP_F32           0
#define EP_TANH_BF16     1
#define EP_TANH_MUL_BF16 2

template <int EP, bool FULL>
__global__ __launch_bounds__(256)
void gemm_bf16_wmma(const unsigned short* __restrict__ A,
                    const unsigned short* __restrict__ B,
                    int Mrows, int Ncols, int K,
                    const float* __restrict__ bias,
                    const float* __restrict__ mul,
                    float* __restrict__ Cf,
                    unsigned short* __restrict__ Cbf) {
  const int BM = 128, BN = 128, BK = 32, LDT = 40;  // LDT padded vs 32 to kill bank conflicts
  __shared__ __align__(16) unsigned short As[2][BM * LDT];
  __shared__ __align__(16) unsigned short Bs[2][BN * LDT];

  const int tid   = threadIdx.x;
  const int mBase = blockIdx.x * BM;
  const int nBase = blockIdx.y * BN;
  const int wave  = tid >> 5;
  const int lane  = tid & 31;
  const int wm    = wave & 3;       // M group: 0..3 (32 rows each)
  const int wn    = wave >> 2;      // N group: 0..1 (64 cols each)
  const int r     = lane & 15;      // row within 16x16 fragment
  const int half  = lane >> 4;      // K-half selector per ISA 16-bit A/B layout
  const int kh    = half * 8;

  // Per-thread staging chunks: 512 chunks of 8 bf16 per 128x32 tile, 2 per thread per matrix.
  const int c0row = tid >> 2;              // chunk 0 row (0..63)
  const int c1row = (tid + 256) >> 2;      // chunk 1 row (64..127)
  const int ckc   = (tid & 3) * 8;         // 0,8,16,24

  auto issue_tile = [&](int k0, int buf) {
    if (FULL) {
      // All guards statically true: 4 straight-line async DMAs, full EXEC.
      async_copy_b128(lds_u32(&As[buf][c0row * LDT + ckc]), A + (long)(mBase + c0row) * K + k0 + ckc);
      async_copy_b128(lds_u32(&As[buf][c1row * LDT + ckc]), A + (long)(mBase + c1row) * K + k0 + ckc);
      async_copy_b128(lds_u32(&Bs[buf][c0row * LDT + ckc]), B + (long)(nBase + c0row) * K + k0 + ckc);
      async_copy_b128(lds_u32(&Bs[buf][c1row * LDT + ckc]), B + (long)(nBase + c1row) * K + k0 + ckc);
    } else {
      int gr = mBase + c0row;
      if (gr < Mrows)
        async_copy_b128(lds_u32(&As[buf][c0row * LDT + ckc]), A + (long)gr * K + k0 + ckc);
      gr = mBase + c1row;
      if (gr < Mrows)
        async_copy_b128(lds_u32(&As[buf][c1row * LDT + ckc]), A + (long)gr * K + k0 + ckc);
      int gc = nBase + c0row;
      if (gc < Ncols)
        async_copy_b128(lds_u32(&Bs[buf][c0row * LDT + ckc]), B + (long)gc * K + k0 + ckc);
      gc = nBase + c1row;
      if (gc < Ncols)
        async_copy_b128(lds_u32(&Bs[buf][c1row * LDT + ckc]), B + (long)gc * K + k0 + ckc);
    }
  };

  const v8f zerov = {0.f, 0.f, 0.f, 0.f, 0.f, 0.f, 0.f, 0.f};
  v8f acc[2][4];
#pragma unroll
  for (int i = 0; i < 2; ++i)
#pragma unroll
    for (int j = 0; j < 4; ++j) acc[i][j] = zerov;

  issue_tile(0, 0);
  int cur = 0;

  for (int k0 = 0; k0 < K; k0 += BK) {
    wait_async0();      // this wave's tile-k async copies have landed in LDS
    __syncthreads();    // => every wave's tile-k data visible; everyone done with tile k-1
    if (k0 + BK < K) issue_tile(k0 + BK, cur ^ 1);  // overlaps with the WMMAs below

    // Fragments per ISA 16-bit layout: half 0 -> K {0..7,16..23}, half 1 -> K {8..15,24..31}
    v16bf af[2];
#pragma unroll
    for (int i = 0; i < 2; ++i) {
      const unsigned short* p = &As[cur][(wm * 32 + i * 16 + r) * LDT];
      v8bf lo = *(const v8bf*)(p + kh);
      v8bf hi = *(const v8bf*)(p + 16 + kh);
      af[i] = __builtin_shufflevector(lo, hi, 0, 1, 2, 3, 4, 5, 6, 7, 8, 9, 10, 11, 12, 13, 14, 15);
    }
    v16bf bfrag[4];
#pragma unroll
    for (int j = 0; j < 4; ++j) {
      const unsigned short* p = &Bs[cur][(wn * 64 + j * 16 + r) * LDT];
      v8bf lo = *(const v8bf*)(p + kh);
      v8bf hi = *(const v8bf*)(p + 16 + kh);
      bfrag[j] = __builtin_shufflevector(lo, hi, 0, 1, 2, 3, 4, 5, 6, 7, 8, 9, 10, 11, 12, 13, 14, 15);
    }

#pragma unroll
    for (int i = 0; i < 2; ++i)
#pragma unroll
      for (int j = 0; j < 4; ++j)
        acc[i][j] = __builtin_amdgcn_wmma_f32_16x16x32_bf16(
            false, af[i], false, bfrag[j], (short)0, acc[i][j], false, false);

    cur ^= 1;
  }

  // Epilogue. D layout: VGPR v holds M = v + 8*half, N = lane&15.
#pragma unroll
  for (int i = 0; i < 2; ++i) {
    int rbase = mBase + wm * 32 + i * 16 + half * 8;
#pragma unroll
    for (int j = 0; j < 4; ++j) {
      int cc = nBase + wn * 64 + j * 16 + r;
      if (FULL || cc < Ncols) {
        float bv = bias[cc];
#pragma unroll
        for (int v = 0; v < 8; ++v) {
          int rr = rbase + v;
          if (FULL || rr < Mrows) {
            float val = acc[i][j][v] + bv;
            if (EP == EP_F32) {
              // Streaming output (o_c is 134MB write-once): non-temporal, keep L2 for tiles.
              __builtin_nontemporal_store(val, &Cf[(long)rr * Ncols + cc]);
            } else {
              float tv = tanhf(val);
              if (EP == EP_TANH_MUL_BF16) tv *= mul[(long)rr * Ncols + cc];
              Cbf[(long)rr * Ncols + cc] = f32_to_bf16_rne(tv);
            }
          }
        }
      }
    }
  }
}

// ---------------------------------------------------------------------------
// Host orchestration
// ---------------------------------------------------------------------------
extern "C" void kernel_launch(void* const* d_in, const int* in_sizes, int n_in,
                              void* d_out, int out_size, void* d_ws, size_t ws_size,
                              hipStream_t stream) {
  const float* t      = (const float*)d_in[0];
  const float* AGE    = (const float*)d_in[1];
  const int*   Endx   = (const int*)d_in[2];
  const float* W_age  = (const float*)d_in[3];
  const float* b_age  = (const float*)d_in[4];
  const float* W_prjT = (const float*)d_in[5];
  const float* b_prjT = (const float*)d_in[6];
  const float* W_prjL = (const float*)d_in[7];
  const float* b_prjL = (const float*)d_in[8];
  const float* outW   = (const float*)d_in[9];
  const float* outb   = (const float*)d_in[10];
  const float* W_fc3  = (const float*)d_in[11];
  const float* b_fc3  = (const float*)d_in[12];

  float* o_c = (float*)d_out;                       // [8192 x 4096]
  float* o_f = o_c + (size_t)N_ROWS * M_;           // [8192 x 50]

  char* base = (char*)d_ws;
  size_t off = 0;
  auto alloc = [&](size_t bytes) -> void* {
    void* p = base + off;
    off = (off + bytes + 255) & ~(size_t)255;
    return p;
  };

  unsigned short* t_bf     = (unsigned short*)alloc((size_t)N_ROWS * DIN_ * 2);
  unsigned short* WprjT_bf = (unsigned short*)alloc((size_t)HID_ * DIN_ * 2);
  unsigned short* WprjL_bf = (unsigned short*)alloc((size_t)HID_ * DIN_ * 2);
  unsigned short* Wage_bf  = (unsigned short*)alloc((size_t)DIN_ * DIN_ * 2);
  unsigned short* Wfc3_bf  = (unsigned short*)alloc((size_t)OUTFE_ * HID_ * 2);
  unsigned short* Ag_bf    = (unsigned short*)alloc((size_t)M_ * DIN_ * 2);
  float*          Zg       = (float*)alloc((size_t)M_ * DIN_ * 4);
  unsigned short* Eg_bf    = (unsigned short*)alloc((size_t)M_ * DIN_ * 2);
  unsigned short* h_bf     = (unsigned short*)alloc((size_t)N_ROWS * HID_ * 2);
  unsigned short* P_bf     = (unsigned short*)alloc((size_t)M_ * HID_ * 2);
  (void)ws_size; (void)in_sizes; (void)n_in; (void)out_size;

  // ---- bf16 conversions (cheap, bandwidth-trivial) ----
  conv_f32_to_bf16<<<512, 256, 0, stream>>>(t, t_bf, (long)N_ROWS * DIN_);
  conv_f32_to_bf16<<<128, 256, 0, stream>>>(W_prjT, WprjT_bf, (long)HID_ * DIN_);
  conv_f32_to_bf16<<<128, 256, 0, stream>>>(W_prjL, WprjL_bf, (long)HID_ * DIN_);
  conv_f32_to_bf16<<<256, 256, 0, stream>>>(W_age, Wage_bf, (long)DIN_ * DIN_);
  conv_f32_to_bf16<<<64, 256, 0, stream>>>(W_fc3, Wfc3_bf, (long)OUTFE_ * HID_);
  // Gather-first optimization: lintrans only on the 4096 referenced AGE rows
  // (reference computes it for all 20000 rows; it is row-wise so gather commutes).
  gather_conv_bf16<<<M_, 256, 0, stream>>>(AGE, Endx, Ag_bf);

  dim3 blk(256);

  // h = tanh(t @ W_prjT^T + b_prjT) -> bf16 [8192 x 256]
  gemm_bf16_wmma<EP_TANH_BF16, true><<<dim3(N_ROWS / 128, HID_ / 128), blk, 0, stream>>>(
      t_bf, WprjT_bf, N_ROWS, HID_, DIN_, b_prjT, nullptr, nullptr, h_bf);

  // Zg = AGE_inx[Endx] @ W_age^T + b_age -> f32 [4096 x 768]
  gemm_bf16_wmma<EP_F32, true><<<dim3(M_ / 128, DIN_ / 128), blk, 0, stream>>>(
      Ag_bf, Wage_bf, M_, DIN_, DIN_, b_age, nullptr, Zg, nullptr);

  // Row-wise min-max scale + L2 normalize -> bf16 [4096 x 768]
  lintrans_kernel<<<M_, 256, 0, stream>>>(Zg, Eg_bf);

  // P = tanh(Eg @ W_prjL^T + b_prjL) * outW -> bf16 [4096 x 256]
  gemm_bf16_wmma<EP_TANH_MUL_BF16, true><<<dim3(M_ / 128, HID_ / 128), blk, 0, stream>>>(
      Eg_bf, WprjL_bf, M_, HID_, DIN_, b_prjL, outW, nullptr, P_bf);

  // o_c = h @ P^T + outb -> f32 [8192 x 4096]   (the big, bandwidth-bound GEMM)
  gemm_bf16_wmma<EP_F32, true><<<dim3(N_ROWS / 128, M_ / 128), blk, 0, stream>>>(
      h_bf, P_bf, N_ROWS, M_, HID_, outb, nullptr, o_c, nullptr);

  // o_f = h @ W_fc3^T + b_fc3 -> f32 [8192 x 50] (N guarded inside kernel)
  gemm_bf16_wmma<EP_F32, false><<<dim3(N_ROWS / 128, 1), blk, 0, stream>>>(
      h_bf, Wfc3_bf, N_ROWS, OUTFE_, HID_, b_fc3, nullptr, o_f, nullptr);
}
